// MultiHeadGeometryAttention_83141976916329
// MI455X (gfx1250) — compile-verified, matrix operand order checked by reading, not run on hardware
//
#include <hip/hip_runtime.h>

#define BATCH 2
#define NPTS  2048
#define CDIM  512
#define HEADS 8
#define DH    64
#define MROWS (BATCH * NPTS)   // 4096
#define QKVN  (3 * CDIM)       // 1536

typedef __bf16 bf16;
typedef __attribute__((ext_vector_type(16))) __bf16 bf16x16;
typedef __attribute__((ext_vector_type(8)))  __bf16 bf16x8v;
typedef __attribute__((ext_vector_type(8)))  float  f32x8;

__device__ __forceinline__ f32x8 wmma_bf16(bf16x16 a, bf16x16 b, f32x8 c) {
  return __builtin_amdgcn_wmma_f32_16x16x32_bf16(false, a, false, b, (short)0, c,
                                                 false, false);
}

__device__ __forceinline__ f32x8 zero8() {
  f32x8 z;
#pragma unroll
  for (int e = 0; e < 8; ++e) z[e] = 0.0f;
  return z;
}

// WGP-scope prefetch (scope field = 0 is the encoding default, which per the
// ISA pulls the line into ALL cache levels; __builtin_prefetch only reaches
// SE/SYS scope which skips the WGP caches).  NOTE: no "memory" clobber —
// a prefetch needs no ordering, and a clobber would act as a compiler
// barrier that blocks unrolling and load/WMMA pipelining.
__device__ __forceinline__ void prefetch_wgp(const void* p) {
  asm volatile("global_prefetch_b8 %0, off" ::"v"(p));
}

// Load a 16x32 bf16 A/B fragment from a row-major tile (row stride ld elems).
// ISA layout: lanes 0-15 = rows 0-15 holding K 0-7 & 16-23; lanes 16-31 same
// rows holding K 8-15 & 24-31.  Two 16-byte loads per lane.
__device__ __forceinline__ bf16x16 load_frag(const bf16* tile, int ld) {
  const int lane = threadIdx.x & 31;
  const int row  = lane & 15;
  const int koff = (lane & 16) ? 8 : 0;
  const bf16* p = tile + (size_t)row * ld + koff;
  bf16x8v lo = *reinterpret_cast<const bf16x8v*>(p);
  bf16x8v hi = *reinterpret_cast<const bf16x8v*>(p + 16);
  return __builtin_shufflevector(lo, hi, 0, 1, 2, 3, 4, 5, 6, 7,
                                 8, 9, 10, 11, 12, 13, 14, 15);
}

// ---------------------------------------------------------------- converts
__global__ void cvt_bf16_kernel(const float* __restrict__ src,
                                bf16* __restrict__ dst, int n) {
  int i = blockIdx.x * blockDim.x + threadIdx.x;
  if (i < n) dst[i] = (bf16)src[i];
}

// src[rows][cols] f32 -> dst[cols][rows] bf16
__global__ void transpose_bf16_kernel(const float* __restrict__ src,
                                      bf16* __restrict__ dst, int rows, int cols) {
  int i = blockIdx.x * blockDim.x + threadIdx.x;
  if (i < rows * cols) {
    int r = i / cols, c = i % cols;
    dst[(size_t)c * rows + r] = (bf16)src[i];
  }
}

// -------------------------------------------------------- geometry bias
// bias[b][i][j] = GAMMA * exp(-(aniso2+1e-8)/2) * (n_i.n_j) + curv_j*dens_j*lin_j
// (ALPHA0=BETA0=SIGMA=1, GAMMA=0.5; sqrt elided since w uses aniso^2)
__global__ void geom_bias_kernel(const float* __restrict__ coord,
                                 const float* __restrict__ pdir,
                                 const float* __restrict__ curv,
                                 const float* __restrict__ dens,
                                 const float* __restrict__ nrm,
                                 const float* __restrict__ lin,
                                 float* __restrict__ bias) {
  const int i = blockIdx.x;
  const int b = blockIdx.y;
  const size_t pi = (size_t)b * NPTS + i;
  const float cix = coord[pi * 3 + 0], ciy = coord[pi * 3 + 1], ciz = coord[pi * 3 + 2];
  const float dix = pdir[pi * 3 + 0],  diy = pdir[pi * 3 + 1],  diz = pdir[pi * 3 + 2];
  const float nix = nrm[pi * 3 + 0],   niy = nrm[pi * 3 + 1],   niz = nrm[pi * 3 + 2];
  const float alpha  = lin[pi];
  const float projii = dix * cix + diy * ciy + diz * ciz;
  const float sqi    = cix * cix + ciy * ciy + ciz * ciz;
  float* brow = bias + ((size_t)b * NPTS + i) * NPTS;
  for (int j = threadIdx.x; j < NPTS; j += blockDim.x) {
    const size_t pj = (size_t)b * NPTS + j;
    const float cjx = coord[pj * 3 + 0], cjy = coord[pj * 3 + 1], cjz = coord[pj * 3 + 2];
    const float par  = projii - (dix * cjx + diy * cjy + diz * cjz);
    const float dot  = cix * cjx + ciy * cjy + ciz * cjz;
    const float sqj  = cjx * cjx + cjy * cjy + cjz * cjz;
    const float d2   = sqi + sqj - 2.0f * dot;
    const float par2 = par * par;
    const float perp2  = fmaxf(d2 - par2, 0.0f);
    const float aniso2 = alpha * par2 + perp2;
    const float w      = __expf(-0.5f * (aniso2 + 1e-8f));
    const float align  = nix * nrm[pj * 3 + 0] + niy * nrm[pj * 3 + 1] + niz * nrm[pj * 3 + 2];
    const float featj  = curv[pj] * dens[pj] * lin[pj];
    brow[j] = 0.5f * w * align + featj;
  }
}

// ------------------------------------------------- shared 32x64 WMMA loop
// Wave computes a 32x64 output tile: 8 WMMAs per 6 fragment loads.
// K dimension fixed to CDIM so the trip count is a compile-time constant.
__device__ __forceinline__ void gemm32x64(const bf16* A, const bf16* BT,
                                          int lda, int ldb, f32x8 (&acc)[2][4]) {
  const int lane = threadIdx.x & 31;
#pragma unroll
  for (int k0 = 0; k0 < CDIM; k0 += 32) {
    if (k0 + 32 < CDIM) {   // WGP-scope prefetch of next K panel (L2-hot)
      prefetch_wgp(A + (size_t)lane * lda + k0 + 32);
      prefetch_wgp(BT + (size_t)lane * ldb + k0 + 32);
      prefetch_wgp(BT + (size_t)(32 + lane) * ldb + k0 + 32);
    }
    bf16x16 a0 = load_frag(A + k0, lda);
    bf16x16 a1 = load_frag(A + (size_t)16 * lda + k0, lda);
#pragma unroll
    for (int j = 0; j < 4; ++j) {
      bf16x16 bj = load_frag(BT + (size_t)(16 * j) * ldb + k0, ldb);
      acc[0][j] = wmma_bf16(a0, bj, acc[0][j]);
      acc[1][j] = wmma_bf16(a1, bj, acc[1][j]);
    }
  }
}

// ------------------------------------------------------------- QKV GEMM
// qkv = x @ Wqkv + bqkv; scatter into q[B,H,N,Dh] (scaled), k[B,H,N,Dh], vT[B,H,Dh,N]
__global__ void qkv_gemm_kernel(const bf16* __restrict__ xb,
                                const bf16* __restrict__ WT,   // [1536][512]
                                const float* __restrict__ bqkv,
                                bf16* __restrict__ q, bf16* __restrict__ k,
                                bf16* __restrict__ vT) {
  const int widx = threadIdx.x >> 5, lane = threadIdx.x & 31;
  const int gw = blockIdx.x * 4 + widx;
  const int TN = QKVN / 64;                       // 24 col tiles
  const int row0 = (gw / TN) * 32, col0 = (gw % TN) * 64;
  f32x8 acc[2][4];
#pragma unroll
  for (int a = 0; a < 2; ++a)
#pragma unroll
    for (int c = 0; c < 4; ++c) acc[a][c] = zero8();
  gemm32x64(xb + (size_t)row0 * CDIM, WT + (size_t)col0 * CDIM, CDIM, CDIM, acc);
  const int colL = lane & 15, rbase = (lane & 16) ? 8 : 0;
#pragma unroll
  for (int ti = 0; ti < 2; ++ti)
#pragma unroll
    for (int tj = 0; tj < 4; ++tj)
#pragma unroll
      for (int r = 0; r < 8; ++r) {
        const int m = row0 + ti * 16 + rbase + r;
        const int n = col0 + tj * 16 + colL;
        const float v = acc[ti][tj][r] + bqkv[n];
        const int which = n / CDIM, c = n % CDIM;
        const int hh = c >> 6, d = c & 63;
        const int bb = m >> 11, i = m & (NPTS - 1);
        const size_t hoff = (size_t)bb * HEADS + hh;
        if (which == 0)      q[(hoff * NPTS + i) * DH + d] = (bf16)(v * 0.125f);
        else if (which == 1) k[(hoff * NPTS + i) * DH + d] = (bf16)v;
        else                 vT[(hoff * DH + d) * NPTS + i] = (bf16)v;
      }
}

// --------------------------------------------------- flash attention w/ bias
// One wave owns 32 query rows (two 16-row subtiles): K fragments feed 8 S-WMMAs,
// each V fragment feeds 2 O-WMMAs -> halves L2 traffic vs 16-row waves.
__global__ void attn_kernel(const bf16* __restrict__ q, const bf16* __restrict__ k,
                            const bf16* __restrict__ vT,
                            const float* __restrict__ bias,
                            bf16* __restrict__ attn_out) {
  __shared__ alignas(16) bf16 ldsP[4][2][16 * 32];
  const int widx = threadIdx.x >> 5, lane = threadIdx.x & 31;
  const int gw = blockIdx.x * 4 + widx;           // 1024 waves total
  const int itile = gw & 63;                      // N/32 row tiles
  const int bh = gw >> 6;                         // 0..15
  const int bb = bh >> 3, hh = bh & 7;
  const int i0 = itile * 32;
  const bf16* qh = q  + (size_t)bh * NPTS * DH;
  const bf16* kh = k  + (size_t)bh * NPTS * DH;
  const bf16* vh = vT + (size_t)bh * DH * NPTS;
  const float* biasb = bias + (size_t)bb * NPTS * NPTS;

  bf16x16 aq[2][2];
#pragma unroll
  for (int u = 0; u < 2; ++u) {
    aq[u][0] = load_frag(qh + (size_t)(i0 + 16 * u) * DH + 0,  DH);
    aq[u][1] = load_frag(qh + (size_t)(i0 + 16 * u) * DH + 32, DH);
  }

  float mrow[2][8], lrow[2][8];
  f32x8 o[2][4];
#pragma unroll
  for (int u = 0; u < 2; ++u) {
#pragma unroll
    for (int r = 0; r < 8; ++r) { mrow[u][r] = -1e30f; lrow[u][r] = 0.0f; }
#pragma unroll
    for (int t = 0; t < 4; ++t) o[u][t] = zero8();
  }

  const int col = lane & 15;
  const int rbase = (lane & 16) ? 8 : 0;

  for (int j0 = 0; j0 < NPTS; j0 += 32) {
    if (j0 + 32 < NPTS) {   // WGP-scope prefetch of next j-block (K, V^T, bias)
      prefetch_wgp(kh + (size_t)(j0 + 32 + lane) * DH);
      prefetch_wgp(vh + (size_t)lane * NPTS + j0 + 32);
      prefetch_wgp(vh + (size_t)(32 + lane) * NPTS + j0 + 32);
      prefetch_wgp(biasb + (size_t)(i0 + lane) * NPTS + j0 + 32);
    }
    bf16x16 bk[4];
    bk[0] = load_frag(kh + (size_t)(j0)      * DH + 0,  DH);
    bk[1] = load_frag(kh + (size_t)(j0)      * DH + 32, DH);
    bk[2] = load_frag(kh + (size_t)(j0 + 16) * DH + 0,  DH);
    bk[3] = load_frag(kh + (size_t)(j0 + 16) * DH + 32, DH);

#pragma unroll
    for (int u = 0; u < 2; ++u) {
      // C operand preloaded with geometry bias (q pre-scaled by Dh^-0.5)
      f32x8 sA, sB;
      const float* bp = biasb + (size_t)(i0 + 16 * u + rbase) * NPTS + j0 + col;
#pragma unroll
      for (int r = 0; r < 8; ++r) {
        sA[r] = bp[(size_t)r * NPTS];
        sB[r] = bp[(size_t)r * NPTS + 16];
      }
      sA = wmma_bf16(aq[u][0], bk[0], sA);
      sA = wmma_bf16(aq[u][1], bk[1], sA);
      sB = wmma_bf16(aq[u][0], bk[2], sB);
      sB = wmma_bf16(aq[u][1], bk[3], sB);

      // online softmax: row spread over 16 lanes (same half), 8 rows per lane
#pragma unroll
      for (int r = 0; r < 8; ++r) {
        float t = fmaxf(sA[r], sB[r]);
        t = fmaxf(t, __shfl_xor(t, 1, 16));
        t = fmaxf(t, __shfl_xor(t, 2, 16));
        t = fmaxf(t, __shfl_xor(t, 4, 16));
        t = fmaxf(t, __shfl_xor(t, 8, 16));
        const float mnew = fmaxf(mrow[u][r], t);
        const float corr = __expf(mrow[u][r] - mnew);
        mrow[u][r] = mnew;
        const float pa = __expf(sA[r] - mnew);
        const float pb = __expf(sB[r] - mnew);
        float ps = pa + pb;
        ps += __shfl_xor(ps, 1, 16);
        ps += __shfl_xor(ps, 2, 16);
        ps += __shfl_xor(ps, 4, 16);
        ps += __shfl_xor(ps, 8, 16);
        lrow[u][r] = lrow[u][r] * corr + ps;
#pragma unroll
        for (int t4 = 0; t4 < 4; ++t4) o[u][t4][r] *= corr;
        // D-layout -> LDS (16x32 row-major) for A-fragment reload
        ldsP[widx][u][(r + rbase) * 32 + col]      = (bf16)pa;
        ldsP[widx][u][(r + rbase) * 32 + col + 16] = (bf16)pb;
      }
    }
    const bf16x16 pf0 = load_frag(&ldsP[widx][0][0], 32);
    const bf16x16 pf1 = load_frag(&ldsP[widx][1][0], 32);
#pragma unroll
    for (int t4 = 0; t4 < 4; ++t4) {
      const bf16x16 bv = load_frag(vh + (size_t)(t4 * 16) * NPTS + j0, NPTS);
      o[0][t4] = wmma_bf16(pf0, bv, o[0][t4]);
      o[1][t4] = wmma_bf16(pf1, bv, o[1][t4]);
    }
  }
  // finalize: divide by row sum, write bf16 into [B,N,C] layout
#pragma unroll
  for (int u = 0; u < 2; ++u)
#pragma unroll
    for (int r = 0; r < 8; ++r) {
      const float inv = 1.0f / lrow[u][r];
      const int row = i0 + 16 * u + rbase + r;
      bf16* op = attn_out + ((size_t)bb * NPTS + row) * CDIM + hh * DH + col;
#pragma unroll
      for (int t4 = 0; t4 < 4; ++t4) op[t4 * 16] = (bf16)(o[u][t4][r] * inv);
    }
}

// ------------------------------------- gate GEMM: y = sigmoid(x@Wg+bg) * attn
__global__ void gate_gemm_kernel(const bf16* __restrict__ xb,
                                 const bf16* __restrict__ WgT,
                                 const float* __restrict__ bg,
                                 const bf16* __restrict__ attn,
                                 bf16* __restrict__ y) {
  const int widx = threadIdx.x >> 5, lane = threadIdx.x & 31;
  const int gw = blockIdx.x * 4 + widx;
  const int TN = CDIM / 64;                        // 8
  const int row0 = (gw / TN) * 32, col0 = (gw % TN) * 64;
  f32x8 acc[2][4];
#pragma unroll
  for (int a = 0; a < 2; ++a)
#pragma unroll
    for (int c = 0; c < 4; ++c) acc[a][c] = zero8();
  gemm32x64(xb + (size_t)row0 * CDIM, WgT + (size_t)col0 * CDIM, CDIM, CDIM, acc);
  const int colL = lane & 15, rbase = (lane & 16) ? 8 : 0;
#pragma unroll
  for (int ti = 0; ti < 2; ++ti)
#pragma unroll
    for (int tj = 0; tj < 4; ++tj)
#pragma unroll
      for (int r = 0; r < 8; ++r) {
        const int m = row0 + ti * 16 + rbase + r;
        const int n = col0 + tj * 16 + colL;
        float g = acc[ti][tj][r] + bg[n];
        g = 1.0f / (1.0f + __expf(-g));
        y[(size_t)m * CDIM + n] = (bf16)(g * (float)attn[(size_t)m * CDIM + n]);
      }
}

// ---------------------------------------- output GEMM: out = y@Wout + bout
__global__ void out_gemm_kernel(const bf16* __restrict__ yb,
                                const bf16* __restrict__ WoT,
                                const float* __restrict__ bo,
                                float* __restrict__ out) {
  const int widx = threadIdx.x >> 5, lane = threadIdx.x & 31;
  const int gw = blockIdx.x * 4 + widx;
  const int TN = CDIM / 64;
  const int row0 = (gw / TN) * 32, col0 = (gw % TN) * 64;
  f32x8 acc[2][4];
#pragma unroll
  for (int a = 0; a < 2; ++a)
#pragma unroll
    for (int c = 0; c < 4; ++c) acc[a][c] = zero8();
  gemm32x64(yb + (size_t)row0 * CDIM, WoT + (size_t)col0 * CDIM, CDIM, CDIM, acc);
  const int colL = lane & 15, rbase = (lane & 16) ? 8 : 0;
#pragma unroll
  for (int ti = 0; ti < 2; ++ti)
#pragma unroll
    for (int tj = 0; tj < 4; ++tj)
#pragma unroll
      for (int r = 0; r < 8; ++r) {
        const int m = row0 + ti * 16 + rbase + r;
        const int n = col0 + tj * 16 + colL;
        out[(size_t)m * CDIM + n] = acc[ti][tj][r] + bo[n];
      }
}

// ------------------------------------------------------------------ launch
extern "C" void kernel_launch(void* const* d_in, const int* in_sizes, int n_in,
                              void* d_out, int out_size, void* d_ws, size_t ws_size,
                              hipStream_t stream) {
  (void)in_sizes; (void)n_in; (void)out_size; (void)ws_size;
  const float* x     = (const float*)d_in[0];
  const float* coord = (const float*)d_in[1];
  const float* pdir  = (const float*)d_in[2];
  const float* curv  = (const float*)d_in[3];
  const float* dens  = (const float*)d_in[4];
  const float* nrm   = (const float*)d_in[5];
  const float* lin   = (const float*)d_in[6];
  const float* Wqkv  = (const float*)d_in[7];
  const float* bqkv  = (const float*)d_in[8];
  const float* Wout  = (const float*)d_in[9];
  const float* bout  = (const float*)d_in[10];
  const float* Wgate = (const float*)d_in[11];
  const float* bgate = (const float*)d_in[12];

  char* ws = (char*)d_ws;
  size_t off = 0;
  auto alloc = [&](size_t bytes) -> char* {
    char* p = ws + off;
    off += (bytes + 255) & ~(size_t)255;
    return p;
  };
  bf16* xb     = (bf16*)alloc((size_t)MROWS * CDIM * sizeof(bf16));
  bf16* WqkvT  = (bf16*)alloc((size_t)QKVN * CDIM * sizeof(bf16));
  bf16* WgateT = (bf16*)alloc((size_t)CDIM * CDIM * sizeof(bf16));
  bf16* WoutT  = (bf16*)alloc((size_t)CDIM * CDIM * sizeof(bf16));
  bf16* qb     = (bf16*)alloc((size_t)BATCH * HEADS * NPTS * DH * sizeof(bf16));
  bf16* kb     = (bf16*)alloc((size_t)BATCH * HEADS * NPTS * DH * sizeof(bf16));
  bf16* vTb    = (bf16*)alloc((size_t)BATCH * HEADS * DH * NPTS * sizeof(bf16));
  bf16* attnb  = (bf16*)alloc((size_t)MROWS * CDIM * sizeof(bf16));
  bf16* yb     = (bf16*)alloc((size_t)MROWS * CDIM * sizeof(bf16));
  float* biasb = (float*)alloc((size_t)BATCH * NPTS * NPTS * sizeof(float));

  const int nx = MROWS * CDIM;
  cvt_bf16_kernel<<<(nx + 255) / 256, 256, 0, stream>>>(x, xb, nx);
  transpose_bf16_kernel<<<(CDIM * QKVN + 255) / 256, 256, 0, stream>>>(Wqkv, WqkvT, CDIM, QKVN);
  transpose_bf16_kernel<<<(CDIM * CDIM + 255) / 256, 256, 0, stream>>>(Wgate, WgateT, CDIM, CDIM);
  transpose_bf16_kernel<<<(CDIM * CDIM + 255) / 256, 256, 0, stream>>>(Wout, WoutT, CDIM, CDIM);

  geom_bias_kernel<<<dim3(NPTS, BATCH), 256, 0, stream>>>(coord, pdir, curv, dens,
                                                          nrm, lin, biasb);

  // waves = (M/32)*(QKVN/64) = 128*24 = 3072 -> 768 blocks of 4 waves
  qkv_gemm_kernel<<<3072 / 4, 128, 0, stream>>>(xb, WqkvT, bqkv, qb, kb, vTb);

  // waves = B*H*(N/32) = 1024 -> 256 blocks of 4 waves
  attn_kernel<<<1024 / 4, 128, 0, stream>>>(qb, kb, vTb, biasb, attnb);

  // waves = (M/32)*(CDIM/64) = 128*8 = 1024 -> 256 blocks
  gate_gemm_kernel<<<1024 / 4, 128, 0, stream>>>(xb, WgateT, bgate, attnb, yb);
  out_gemm_kernel<<<1024 / 4, 128, 0, stream>>>(yb, WoutT, bout, (float*)d_out);
}
